// GCN_Kron_66958540144870
// MI455X (gfx1250) — matrix operand order repaired and checked by reading, not compiled, not run on hardware
//
#include <hip/hip_runtime.h>
#include <cstdint>
#include <cstddef>

typedef __attribute__((ext_vector_type(4)))  float  vf4;
typedef __attribute__((ext_vector_type(8)))  float  v8f;
typedef __attribute__((ext_vector_type(16))) __bf16 v16bf;

#define NN 10000
#define NP 10016  // N padded to a multiple of 32 (zero-filled pad)

// -------- convert x (N x 64, row-major fp32) -> transposed bf16 hi/lo --------
__global__ __launch_bounds__(256)
void cvt_xT_kernel(const float* __restrict__ x, __bf16* __restrict__ hiT,
                   __bf16* __restrict__ loT) {
  int tid = blockIdx.x * blockDim.x + threadIdx.x;
  if (tid >= 64 * NP) return;
  int f = tid / NP;
  int n = tid - f * NP;
  float v = (n < NN) ? x[(size_t)n * 64 + f] : 0.0f;
  __bf16 hi = (__bf16)v;
  __bf16 lo = (__bf16)(v - (float)hi);
  hiT[(size_t)f * NP + n] = hi;
  loT[(size_t)f * NP + n] = lo;
}

// -------- T = A @ H  via v_wmma_f32_16x16x32_bf16, hi/lo split of H --------
// A: NN x NN fp32 row-major (exact 0/1 -> exact in bf16)
// HhiT/HloT: [F][NP] bf16 (column slices of H are contiguous)
// One block = 4 waves; each wave computes a 16-row strip x all F columns.
template <int F>
__global__ __launch_bounds__(128)
void gemm_AH(const float* __restrict__ A, const __bf16* __restrict__ Hhi,
             const __bf16* __restrict__ Hlo, float* __restrict__ T) {
  constexpr int NT = F / 16;
  __shared__ __align__(32) __bf16 sh_hi[F * 32];
  __shared__ __align__(32) __bf16 sh_lo[F * 32];

  int strip = blockIdx.x * 4 + (threadIdx.x >> 5);
  if (strip > (NN / 16) - 1) strip = (NN / 16) - 1;  // duplicate tail strip (same data)
  int l    = threadIdx.x & 31;
  int m    = l & 15;   // A row within strip / B column within tile
  int half = l >> 4;   // lane half per WMMA operand layout
  const float* Arow = A + (size_t)(strip * 16 + m) * NN;

  v8f z8 = {0.f, 0.f, 0.f, 0.f, 0.f, 0.f, 0.f, 0.f};
  v8f acc[NT];
#pragma unroll
  for (int t = 0; t < NT; ++t) acc[t] = z8;

  for (int k0 = 0; k0 < NN; k0 += 32) {
    __syncthreads();
    // Cooperative stage of B tile (hi & lo): F columns x 32 K-values each.
    for (int c = threadIdx.x; c < F * 4; c += 128) {
      int col = c >> 2, part = c & 3;
      size_t src = (size_t)col * NP + (size_t)k0 + (size_t)part * 8;
      *(uint4*)(sh_hi + c * 8) = *(const uint4*)(Hhi + src);
      *(uint4*)(sh_lo + c * 8) = *(const uint4*)(Hlo + src);
    }
    __syncthreads();

    // A operand 16x32, contiguous k order (k = 16*half + s), same as B.
    // lane half 0 holds K = k0+0..15 ; half 1 holds K = k0+16..31
    int kb1 = k0 + half * 16;
    int kb2 = k0 + half * 16 + 8;
    vf4 u0 = {0.f, 0.f, 0.f, 0.f}, u1 = u0, u2 = u0, u3 = u0;
    if (kb1 + 8 <= NN) { const vf4* p = (const vf4*)(Arow + kb1); u0 = p[0]; u1 = p[1]; }
    if (kb2 + 8 <= NN) { const vf4* p = (const vf4*)(Arow + kb2); u2 = p[0]; u3 = p[1]; }
    if (k0 + 544 <= NN) __builtin_prefetch(Arow + k0 + 512, 0, 1);

    v16bf aop;
#pragma unroll
    for (int i = 0; i < 4; ++i) {
      aop[i]      = (__bf16)u0[i];
      aop[4 + i]  = (__bf16)u1[i];
      aop[8 + i]  = (__bf16)u2[i];
      aop[12 + i] = (__bf16)u3[i];
    }

#pragma unroll
    for (int t = 0; t < NT; ++t) {
      int col = t * 16 + m;  // B: lane n = column, halves hold K 0..15 / 16..31
      v16bf bhi = *(const v16bf*)(sh_hi + col * 32 + half * 16);
      v16bf blo = *(const v16bf*)(sh_lo + col * 32 + half * 16);
      acc[t] = __builtin_amdgcn_wmma_f32_16x16x32_bf16(false, aop, false, bhi,
                                                       (short)0, acc[t], false, false);
      acc[t] = __builtin_amdgcn_wmma_f32_16x16x32_bf16(false, aop, false, blo,
                                                       (short)0, acc[t], false, false);
    }
  }

  // C/D layout: VGPR r -> lanes 0-15: M=r, N=lane ; lanes 16-31: M=8+r, N=lane-16
#pragma unroll
  for (int t = 0; t < NT; ++t) {
    int col = t * 16 + m;
#pragma unroll
    for (int r = 0; r < 8; ++r) {
      int row = strip * 16 + half * 8 + r;
      T[(size_t)row * F + col] = acc[t][r];
    }
  }
}

// -------- a = t @ W ; h = act(a); write a (and z), next-h transposed hi/lo --------
// ACT: 0=relu 1=silu 2=elu 3=leaky 4=sigmoid(final: writes z, no hiT/loT)
template <int FIN, int FOUT, int ACT>
__global__ __launch_bounds__(256)
void layer_kernel(const float* __restrict__ T, const float* __restrict__ W,
                  float* __restrict__ Aout, float* __restrict__ Zout,
                  __bf16* __restrict__ hiT, __bf16* __restrict__ loT) {
  __shared__ float w[FIN * FOUT];
  for (int i = threadIdx.x; i < FIN * FOUT; i += 256) w[i] = W[i];
  __syncthreads();

  int tid = blockIdx.x * blockDim.x + threadIdx.x;
  if (tid >= NP * FOUT) return;
  int n = tid / FOUT;
  int j = tid - n * FOUT;

  if (n >= NN) {  // zero-fill the K padding of the transposed h buffers
    if (ACT != 4) {
      hiT[(size_t)j * NP + n] = (__bf16)0.0f;
      loT[(size_t)j * NP + n] = (__bf16)0.0f;
    }
    return;
  }

  const float* trow = T + (size_t)n * FIN;
  float accv = 0.0f;
#pragma unroll
  for (int k = 0; k < FIN; ++k) accv += trow[k] * w[k * FOUT + j];
  Aout[(size_t)n * FOUT + j] = accv;

  float h;
  if (ACT == 0)      h = fmaxf(accv, 0.0f);
  else if (ACT == 1) h = accv / (1.0f + expf(-accv));
  else if (ACT == 2) h = (accv >= 0.0f) ? accv : (expf(accv) - 1.0f);
  else if (ACT == 3) h = (accv >= 0.0f) ? accv : 0.01f * accv;
  else               h = 1.0f / (1.0f + expf(-accv));

  if (ACT == 4) {
    Zout[(size_t)n * FOUT + j] = h;
  } else {
    __bf16 hi = (__bf16)h;
    __bf16 lo = (__bf16)(h - (float)hi);
    hiT[(size_t)j * NP + n] = hi;
    loT[(size_t)j * NP + n] = lo;
  }
}

extern "C" void kernel_launch(void* const* d_in, const int* in_sizes, int n_in,
                              void* d_out, int out_size, void* d_ws, size_t ws_size,
                              hipStream_t stream) {
  const float* x  = (const float*)d_in[0];
  const float* A  = (const float*)d_in[1];
  const float* W0 = (const float*)d_in[2];
  const float* W1 = (const float*)d_in[3];
  const float* W2 = (const float*)d_in[4];
  const float* W3 = (const float*)d_in[5];
  const float* W4 = (const float*)d_in[6];

  float* out = (float*)d_out;
  float* t1 = out;            // 10000x64
  float* t2 = out +  640000;  // 10000x16
  float* t3 = out +  800000;  // 10000x32
  float* t4 = out + 1120000;  // 10000x16
  float* t5 = out + 1280000;  // 10000x32
  float* a1 = out + 1600000;  // 10000x16
  float* a2 = out + 1760000;  // 10000x32
  float* a3 = out + 2080000;  // 10000x16
  float* a4 = out + 2240000;  // 10000x32
  float* a5 = out + 2560000;  // 10000x8
  float* zz = out + 2640000;  // 10000x8

  // ping-pong transposed-h buffers (bf16 hi/lo), each sized for max F=64
  __bf16* b0hi = (__bf16*)d_ws;
  __bf16* b0lo = b0hi + (size_t)64 * NP;
  __bf16* b1hi = b0lo + (size_t)64 * NP;
  __bf16* b1lo = b1hi + (size_t)64 * NP;

  const int GB = ((NN / 16) + 3) / 4;  // gemm blocks: 4 strips per block
  auto LG = [](int fout) { return (NP * fout + 255) / 256; };

  cvt_xT_kernel<<<(64 * NP + 255) / 256, 256, 0, stream>>>(x, b0hi, b0lo);

  gemm_AH<64><<<GB, 128, 0, stream>>>(A, b0hi, b0lo, t1);
  layer_kernel<64, 16, 0><<<LG(16), 256, 0, stream>>>(t1, W0, a1, nullptr, b1hi, b1lo);

  gemm_AH<16><<<GB, 128, 0, stream>>>(A, b1hi, b1lo, t2);
  layer_kernel<16, 32, 1><<<LG(32), 256, 0, stream>>>(t2, W1, a2, nullptr, b0hi, b0lo);

  gemm_AH<32><<<GB, 128, 0, stream>>>(A, b0hi, b0lo, t3);
  layer_kernel<32, 16, 2><<<LG(16), 256, 0, stream>>>(t3, W2, a3, nullptr, b1hi, b1lo);

  gemm_AH<16><<<GB, 128, 0, stream>>>(A, b1hi, b1lo, t4);
  layer_kernel<16, 32, 3><<<LG(32), 256, 0, stream>>>(t4, W3, a4, nullptr, b0hi, b0lo);

  gemm_AH<32><<<GB, 128, 0, stream>>>(A, b0hi, b0lo, t5);
  layer_kernel<32, 8, 4><<<LG(8), 256, 0, stream>>>(t5, W4, a5, zz, nullptr, nullptr);
}
